// ImprovedCalculateAttention_7722351198476
// MI455X (gfx1250) — compile-verified, matrix operand order checked by reading, not run on hardware
//
#include <hip/hip_runtime.h>

typedef __attribute__((ext_vector_type(16))) _Float16 v16h;
typedef __attribute__((ext_vector_type(8)))  float    v8f;
typedef __attribute__((ext_vector_type(4)))  _Float16 v4h;

constexpr int CB = 2, CH = 8, CS = 2048, CD = 64;
constexpr int TQ  = 128;   // queries per workgroup
constexpr int PB  = 64;    // keys per softmax/WMMA block step (P-tile width)
constexpr int NW  = 8;     // waves per workgroup
constexpr int BKA = 128;   // sync-staging key block (fallback kernel)
constexpr int BKB = 64;    // async-staging key block (double-buffered)
constexpr size_t HKV = (size_t)CB * CH * CS * CD;   // elems per tensor (2M)

// A/B-fragment K-striping for 16-bit 16x32 operands (CDNA5 ISA 7.12.2)
__device__ __forceinline__ int kpat(int v, int hi) {
    return (v < 4 ? 2 * v : 16 + 2 * (v - 4)) + 8 * hi;
}

// generic (shared) pointer -> 32-bit LDS byte offset
__device__ __forceinline__ unsigned lds_u32(const void* p) {
    return (unsigned)(unsigned long long)(__attribute__((address_space(3))) const void*)p;
}

// one lane-wide async 16B copy: LDS[lds] = MEM[gaddr]  (ASYNCcnt-tracked)
__device__ __forceinline__ void async_b128(unsigned lds, unsigned long long gaddr) {
    asm volatile("global_load_async_to_lds_b128 %0, %1, off"
                 :: "v"(lds), "v"(gaddr) : "memory");
}

struct WaveAcc {
    v16h aqx[2], aqy[2];   // Q fragments (pre-scaled by 0.0625*log2e), x and y
    v8f  ox[4], oy[4];     // output accumulators, 4 column tiles of D
    v8f  rs;               // running softmax denominator = P @ ones (matrix-pipe rowsum)
    float m[8];            // online-softmax row max (8 row slots per half-wave)
};

__device__ __forceinline__ void acc_init(WaveAcc& w, const float* qxr, const float* qyr, int hi) {
    const float sc = 0.0625f * 1.44269504088896f;   // mean + 1/sqrt(D) + log2(e), folded into Q
#pragma unroll
    for (int ks = 0; ks < 2; ++ks)
#pragma unroll
        for (int v = 0; v < 8; ++v) {
            const int k = ks * 32 + kpat(v, hi);
            w.aqx[ks][2 * v]     = (_Float16)(qxr[k] * sc);
            w.aqx[ks][2 * v + 1] = (_Float16)(qxr[k + 1] * sc);
            w.aqy[ks][2 * v]     = (_Float16)(qyr[k] * sc);
            w.aqy[ks][2 * v + 1] = (_Float16)(qyr[k + 1] * sc);
        }
    const v8f z = {0.f, 0.f, 0.f, 0.f, 0.f, 0.f, 0.f, 0.f};
#pragma unroll
    for (int t = 0; t < 4; ++t) { w.ox[t] = z; w.oy[t] = z; }
    w.rs = z;
#pragma unroll
    for (int j = 0; j < 8; ++j) w.m[j] = -3.0e38f;
}

// Process 64 keys: scores (16 WMMA), ONE softmax update, rowsum (2 WMMA) + P@V (16 WMMA).
__device__ __forceinline__ void blockstep(WaveAcc& w,
                                          const _Float16* sKx, const _Float16* sKy,
                                          const _Float16* sVx, const _Float16* sVy,
                                          _Float16* sPw, int sb, int ln, int hi) {
    const v8f vzero = {0.f, 0.f, 0.f, 0.f, 0.f, 0.f, 0.f, 0.f};

    // ---- 4 score tiles of 16 keys each (exp2-domain logits; scale folded into Q) ----
    v8f c[4];
#pragma unroll
    for (int t = 0; t < 4; ++t) {
        v16h bx[2], by[2];
        const int kb = (sb + t * 16 + ln) * CD;      // B-frag: lane = N (key)
#pragma unroll
        for (int ks = 0; ks < 2; ++ks)
#pragma unroll
            for (int v = 0; v < 8; ++v) {
                const int k = ks * 32 + kpat(v, hi);
                bx[ks][2 * v]     = sKx[kb + k];
                bx[ks][2 * v + 1] = sKx[kb + k + 1];
                by[ks][2 * v]     = sKy[kb + k];
                by[ks][2 * v + 1] = sKy[kb + k + 1];
            }
        v8f z = vzero;
        z = __builtin_amdgcn_wmma_f32_16x16x32_f16(false, w.aqx[0], false, bx[0], (short)0, z, false, false);
        z = __builtin_amdgcn_wmma_f32_16x16x32_f16(false, w.aqx[1], false, bx[1], (short)0, z, false, false);
        z = __builtin_amdgcn_wmma_f32_16x16x32_f16(false, w.aqy[0], false, by[0], (short)0, z, false, false);
        z = __builtin_amdgcn_wmma_f32_16x16x32_f16(false, w.aqy[1], false, by[1], (short)0, z, false, false);
        c[t] = z;
    }

    // ---- single softmax update for all 64 keys ----
    float mn[8];
#pragma unroll
    for (int j = 0; j < 8; ++j)
        mn[j] = fmaxf(fmaxf(c[0][j], c[1][j]), fmaxf(c[2][j], c[3][j]));
#pragma unroll
    for (int off = 8; off >= 1; off >>= 1)
#pragma unroll
        for (int j = 0; j < 8; ++j) mn[j] = fmaxf(mn[j], __shfl_xor(mn[j], off));

    float alpha[8];
#pragma unroll
    for (int j = 0; j < 8; ++j) {
        const float mN = fmaxf(w.m[j], mn[j]);
        alpha[j] = exp2f(w.m[j] - mN);
        w.m[j] = mN;
    }

    // exponentiate and spill P tile (C-layout -> row-major 16x64) for A-frag reload
#pragma unroll
    for (int t = 0; t < 4; ++t)
#pragma unroll
        for (int j = 0; j < 8; ++j)
            sPw[(j + 8 * hi) * PB + t * 16 + ln] = (_Float16)exp2f(c[t][j] - w.m[j]);

    // rescale running accumulators once per 64 keys
#pragma unroll
    for (int t = 0; t < 4; ++t)
#pragma unroll
        for (int j = 0; j < 8; ++j) { w.ox[t][j] *= alpha[j]; w.oy[t][j] *= alpha[j]; }
#pragma unroll
    for (int j = 0; j < 8; ++j) w.rs[j] *= alpha[j];

    // ---- P back as two A-fragments (16x32 each, K = key dim) ----
    v16h ap[2];
#pragma unroll
    for (int ks = 0; ks < 2; ++ks)
#pragma unroll
        for (int v = 0; v < 8; ++v) {
            const int k = ks * 32 + kpat(v, hi);
            ap[ks][2 * v]     = sPw[ln * PB + k];
            ap[ks][2 * v + 1] = sPw[ln * PB + k + 1];
        }

    // softmax denominator via the matrix pipe: rs += P @ ones  (every column = rowsum)
    v16h pones;
#pragma unroll
    for (int e = 0; e < 16; ++e) pones[e] = (_Float16)1.0f;
    w.rs = __builtin_amdgcn_wmma_f32_16x16x32_f16(false, ap[0], false, pones, (short)0, w.rs, false, false);
    w.rs = __builtin_amdgcn_wmma_f32_16x16x32_f16(false, ap[1], false, pones, (short)0, w.rs, false, false);

    // ---- O += P @ V (4 column tiles x 2 k-steps, for Vx and Vy) ----
#pragma unroll
    for (int t = 0; t < 4; ++t) {
        const int dcol = t * 16 + ln;
#pragma unroll
        for (int ks = 0; ks < 2; ++ks) {
            v16h bvx, bvy;
#pragma unroll
            for (int v = 0; v < 8; ++v) {
                const int k = kpat(v, hi);
                const int r0 = (sb + ks * 32 + k) * CD;
                bvx[2 * v]     = sVx[r0 + dcol];
                bvx[2 * v + 1] = sVx[r0 + CD + dcol];
                bvy[2 * v]     = sVy[r0 + dcol];
                bvy[2 * v + 1] = sVy[r0 + CD + dcol];
            }
            w.ox[t] = __builtin_amdgcn_wmma_f32_16x16x32_f16(false, ap[ks], false, bvx, (short)0, w.ox[t], false, false);
            w.oy[t] = __builtin_amdgcn_wmma_f32_16x16x32_f16(false, ap[ks], false, bvy, (short)0, w.oy[t], false, false);
        }
    }
}

__device__ __forceinline__ void epilogue(const WaveAcc& w, float* O1, float* O2,
                                         size_t base, int qb, int ln, int hi) {
#pragma unroll
    for (int j = 0; j < 8; ++j) {
        const float inv = 1.0f / w.rs[j];
        const size_t row = base + (size_t)(qb + j + 8 * hi) * CD;
#pragma unroll
        for (int t = 0; t < 4; ++t) {
            const int col = t * 16 + ln;
            O1[row + col] = w.ox[t][j] * inv;
            O2[row + col] = w.oy[t][j] * inv;
        }
    }
}

// ---------------- pre-pass: convert K/V tensors f32 -> f16 into workspace ----------------
__global__ __launch_bounds__(256)
void cvt_kv_f16_kernel(const float* __restrict__ Kx, const float* __restrict__ Ky,
                       const float* __restrict__ Vx, const float* __restrict__ Vy,
                       _Float16* __restrict__ w) {
    const size_t i = ((size_t)blockIdx.x * 256 + threadIdx.x) * 4;
    if (i >= HKV) return;
    const float4 a = *(const float4*)(Kx + i);
    const float4 b = *(const float4*)(Ky + i);
    const float4 c = *(const float4*)(Vx + i);
    const float4 d = *(const float4*)(Vy + i);
    *(v4h*)(w + 0 * HKV + i) = (v4h){(_Float16)a.x, (_Float16)a.y, (_Float16)a.z, (_Float16)a.w};
    *(v4h*)(w + 1 * HKV + i) = (v4h){(_Float16)b.x, (_Float16)b.y, (_Float16)b.z, (_Float16)b.w};
    *(v4h*)(w + 2 * HKV + i) = (v4h){(_Float16)c.x, (_Float16)c.y, (_Float16)c.z, (_Float16)c.w};
    *(v4h*)(w + 3 * HKV + i) = (v4h){(_Float16)d.x, (_Float16)d.y, (_Float16)d.z, (_Float16)d.w};
}

// ---------------- main path: async double-buffered f16 K/V from workspace ----------------
__global__ __launch_bounds__(256)
void shared_attn_async_kernel(const float* __restrict__ Qx, const float* __restrict__ Qy,
                              const _Float16* __restrict__ wkv,
                              float* __restrict__ O1, float* __restrict__ O2) {
    __shared__ __align__(16) _Float16 sKV[2][4][BKB * CD];   // [buf][Kx,Ky,Vx,Vy]
    __shared__ __align__(16) _Float16 sP[NW][16 * PB];

    const int tid  = threadIdx.x;
    const int wave = tid >> 5;
    const int lane = tid & 31;
    const int ln   = lane & 15;
    const int hi   = lane >> 4;

    const int bh = blockIdx.y;
    const int qb = blockIdx.x * TQ + wave * 16;
    const size_t base = (size_t)bh * CS * CD;

    // wave-specialized staging role: tensor st (0..3), row-half sh (0..1)
    const int st = wave & 3;
    const int sh = wave >> 2;
    const unsigned long long gsrc0 =
        (unsigned long long)(const void*)(wkv + (size_t)st * HKV + base) +
        (unsigned long long)(sh * 32 * CD + lane * 8) * 2ull;     // bytes; lane covers 16B
    const unsigned ldsl0 = lds_u32(&sKV[0][st][sh * 32 * CD]) + (unsigned)lane * 16u;
    const unsigned ldsl1 = lds_u32(&sKV[1][st][sh * 32 * CD]) + (unsigned)lane * 16u;

    WaveAcc w;
    acc_init(w, Qx + base + (size_t)(qb + ln) * CD, Qy + base + (size_t)(qb + ln) * CD, hi);

    constexpr int NBLK = CS / BKB;   // 32
    // prime buffer 0 with block 0
    {
        const unsigned long long g = gsrc0;
#pragma unroll
        for (int i = 0; i < 8; ++i) async_b128(ldsl0 + i * 512u, g + (unsigned long long)i * 512ull);
    }

    for (int blk = 0; blk < NBLK; ++blk) {
        const int buf = blk & 1;
        if (blk + 1 < NBLK) {    // stage next block into the other buffer
            const unsigned long long g = gsrc0 + (unsigned long long)(blk + 1) * (BKB * CD * 2ull);
            const unsigned l0 = (buf == 0) ? ldsl1 : ldsl0;
#pragma unroll
            for (int i = 0; i < 8; ++i) async_b128(l0 + i * 512u, g + (unsigned long long)i * 512ull);
            asm volatile("s_wait_asynccnt 8" ::: "memory");   // current block landed
        } else {
            asm volatile("s_wait_asynccnt 0" ::: "memory");
        }
        __syncthreads();   // publish staged data to all waves

        blockstep(w, sKV[buf][0], sKV[buf][1], sKV[buf][2], sKV[buf][3],
                  &sP[wave][0], 0, ln, hi);

        __syncthreads();   // done reading buf before it is overwritten next iteration
    }

    epilogue(w, O1, O2, base, qb, ln, hi);
}

// ---------------- fallback: sync staging, f32 source (no workspace needed) ----------------
__global__ __launch_bounds__(256)
void shared_attn_sync_kernel(const float* __restrict__ Qx, const float* __restrict__ Kx,
                             const float* __restrict__ Vx, const float* __restrict__ Qy,
                             const float* __restrict__ Ky, const float* __restrict__ Vy,
                             float* __restrict__ O1, float* __restrict__ O2) {
    __shared__ __align__(16) _Float16 sKx[BKA * CD], sKy[BKA * CD], sVx[BKA * CD], sVy[BKA * CD];
    __shared__ __align__(16) _Float16 sP[NW][16 * PB];

    const int tid  = threadIdx.x;
    const int wave = tid >> 5;
    const int lane = tid & 31;
    const int ln   = lane & 15;
    const int hi   = lane >> 4;

    const int bh = blockIdx.y;
    const int qb = blockIdx.x * TQ + wave * 16;
    const size_t base = (size_t)bh * CS * CD;

    WaveAcc w;
    acc_init(w, Qx + base + (size_t)(qb + ln) * CD, Qy + base + (size_t)(qb + ln) * CD, hi);

    for (int kb0 = 0; kb0 < CS; kb0 += BKA) {
        __syncthreads();
        {
            const size_t g = base + (size_t)kb0 * CD;
#pragma unroll
            for (int idx = tid * 4; idx < BKA * CD; idx += 1024) {
                const float4 a = *(const float4*)(Kx + g + idx);
                const float4 b = *(const float4*)(Ky + g + idx);
                const float4 c = *(const float4*)(Vx + g + idx);
                const float4 d = *(const float4*)(Vy + g + idx);
                *(v4h*)(sKx + idx) = (v4h){(_Float16)a.x, (_Float16)a.y, (_Float16)a.z, (_Float16)a.w};
                *(v4h*)(sKy + idx) = (v4h){(_Float16)b.x, (_Float16)b.y, (_Float16)b.z, (_Float16)b.w};
                *(v4h*)(sVx + idx) = (v4h){(_Float16)c.x, (_Float16)c.y, (_Float16)c.z, (_Float16)c.w};
                *(v4h*)(sVy + idx) = (v4h){(_Float16)d.x, (_Float16)d.y, (_Float16)d.z, (_Float16)d.w};
            }
            if (kb0 + BKA < CS) {
                const size_t gn = g + (size_t)BKA * CD + (size_t)tid * 32;
                __builtin_prefetch(Kx + gn, 0, 0);
                __builtin_prefetch(Ky + gn, 0, 0);
                __builtin_prefetch(Vx + gn, 0, 0);
                __builtin_prefetch(Vy + gn, 0, 0);
            }
        }
        __syncthreads();
#pragma unroll
        for (int sb = 0; sb < BKA; sb += PB)
            blockstep(w, sKx, sKy, sVx, sVy, &sP[wave][0], sb, ln, hi);
    }

    epilogue(w, O1, O2, base, qb, ln, hi);
}

extern "C" void kernel_launch(void* const* d_in, const int* in_sizes, int n_in,
                              void* d_out, int out_size, void* d_ws, size_t ws_size,
                              hipStream_t stream) {
    const float* Qx = (const float*)d_in[0];
    const float* Kx = (const float*)d_in[1];
    const float* Vx = (const float*)d_in[2];
    const float* Qy = (const float*)d_in[3];
    const float* Ky = (const float*)d_in[4];
    const float* Vy = (const float*)d_in[5];
    float* O1 = (float*)d_out;
    float* O2 = O1 + HKV;

    dim3 grid(CS / TQ, CB * CH);
    const size_t need = 4 * HKV * sizeof(_Float16);   // 16 MB of f16 K/V
    if (ws_size >= need) {
        _Float16* wkv = (_Float16*)d_ws;
        cvt_kv_f16_kernel<<<(int)(HKV / 4 / 256), 256, 0, stream>>>(Kx, Ky, Vx, Vy, wkv);
        shared_attn_async_kernel<<<grid, 256, 0, stream>>>(Qx, Qy, wkv, O1, O2);
    } else {
        shared_attn_sync_kernel<<<grid, 256, 0, stream>>>(Qx, Kx, Vx, Qy, Ky, Vy, O1, O2);
    }
}
